// SIVA_82617990906054
// MI455X (gfx1250) — matrix operand order repaired and checked by reading, not compiled, not run on hardware
//
#include <hip/hip_runtime.h>
#include <hip/hip_bf16.h>
#include <math.h>

// Problem constants (from the reference file)
#define NN  4000
#define EE  50000
#define NGD 12
#define HH  128
#define BDD 128
#define LLAY 2
#define BBG 100
#define HH4 512

typedef __bf16 bf16;
typedef bf16  v16bf __attribute__((ext_vector_type(16)));
typedef float v8f   __attribute__((ext_vector_type(8)));

#define WMMA_BF16(a,b,c) __builtin_amdgcn_wmma_f32_16x16x32_bf16(false,(a),false,(b),(short)0,(c),false,false)

// Branch-free GELU (sigmoid form). v_exp_f32 / v_rcp_f32 are single TRANS ops
// that co-execute with the matrix pipe; no exec-mask churn between WMMAs.
__device__ __forceinline__ float gelu_f(float v){
  float s = __builtin_amdgcn_rcpf(1.0f + __expf(-1.702f * v));
  return v * s;
}

// ---- WMMA fragment helpers (CDNA5 16-bit A 16x32 layout; B pre-transposed N x Kpad) ----

// A fragment: LDS region organized [kstep][lane][16 bf16], each lane reads 32B contiguous.
__device__ __forceinline__ v16bf wmma_ldA(const bf16* lds, int kk, int lane){
  return *(const v16bf*)(lds + (size_t)(kk*32 + lane)*16);
}
// B fragment: Wt is bf16, row-major [N][Kpad]; lane holds column n = nt*16+(lane&15),
// K-range 16*(lane>>4) within the 32-wide K-step -> one contiguous 32B load.
__device__ __forceinline__ v16bf wmma_ldB(const bf16* Wt, int Kpad, int nt, int kk, int lane){
  int n  = nt*16 + (lane & 15);
  int k0 = kk*32 + ((lane >> 4) << 4);
  return *(const v16bf*)(Wt + (size_t)n*Kpad + k0);
}
// Scatter one A-matrix element (m,k) into fragment-ordered LDS.
__device__ __forceinline__ void wmma_stA(bf16* lds, int m, int k, float v){
  int kk = k >> 5, kl = k & 31;
  int la = m + ((kl >> 4) << 4);
  int sl = kl & 15;
  lds[(size_t)(kk*32 + la)*16 + sl] = (bf16)v;
}
// Restage a C-layout accumulator tile (cols nt*16..nt*16+15) as A fragments for the next GEMM.
// C layout: lane holds col n = lane&15, rows m = r + 8*(lane>>4).
__device__ __forceinline__ void stage_c(bf16* lds, int nt, int lane, v8f c){
  int col = nt*16 + (lane & 15);
  #pragma unroll
  for (int r = 0; r < 8; ++r){
    int m = r + ((lane >> 4) << 3);
    wmma_stA(lds, m, col, c[r]);
  }
}

// ------------------------------- small prep kernels -------------------------------

__global__ void zero_k(float* p, long long n){
  long long i = (long long)blockIdx.x * blockDim.x + threadIdx.x;
  if (i < n) p[i] = 0.0f;
}

// Wt[n][k] = (k<K) ? W[k][n] : 0   (fp32 -> bf16, transposed, K padded)
__global__ void wt_k(const float* W, bf16* Wt, int K, int Nn, int Kpad){
  int tid = blockIdx.x * blockDim.x + threadIdx.x;
  if (tid >= Nn * Kpad) return;
  int n = tid / Kpad, k = tid - n * Kpad;
  Wt[tid] = (k < K) ? (bf16)W[(size_t)k * Nn + n] : (bf16)0.0f;
}

// grid_graph[b][k][i] = sum_j R[b][i][j] * grid0[k][j]
__global__ void grid_graph_k(const float* R, const float* grid0, float* gg){
  int tid = blockIdx.x * blockDim.x + threadIdx.x;
  if (tid >= BBG * NGD) return;
  int b = tid / NGD, k = tid - b * NGD;
  const float* Rb = R + (size_t)b * 9;
  const float* g0 = grid0 + (size_t)k * 3;
  #pragma unroll
  for (int i = 0; i < 3; ++i)
    gg[(size_t)tid*3 + i] = Rb[i*3+0]*g0[0] + Rb[i*3+1]*g0[1] + Rb[i*3+2]*g0[2];
}

__global__ void stats_k(const float* pos, const int* batch, float* cnt, float* psum){
  int n = blockIdx.x * blockDim.x + threadIdx.x;
  if (n >= NN) return;
  int b = batch[n];
  atomicAdd(&cnt[b], 1.0f);
  atomicAdd(&psum[b*3+0], pos[n*3+0]);
  atomicAdd(&psum[b*3+1], pos[n*3+1]);
  atomicAdd(&psum[b*3+2], pos[n*3+2]);
}

__global__ void mean_k(const float* cnt, const float* psum, float* meanp){
  int b = blockIdx.x * blockDim.x + threadIdx.x;
  if (b >= BBG) return;
  float c = fmaxf(cnt[b], 1.0f);
  meanp[b*3+0] = psum[b*3+0] / c;
  meanp[b*3+1] = psum[b*3+1] / c;
  meanp[b*3+2] = psum[b*3+2] / c;
}

// grid_node + embedding: x[(n,k),c] = feat(n,k) . W_embed[:,c]
__global__ void embed_k(const float* pos, const float* vel, const float* charges,
                        const int* batch, const float* grid_graph, const float* meanp,
                        const float* W_embed, float* grid_node, float* x){
  int tid = blockIdx.x * blockDim.x + threadIdx.x;
  if (tid >= NN * NGD) return;
  int n = tid / NGD, k = tid - n * NGD;
  int b = batch[n];
  float g0 = grid_graph[((size_t)b*NGD + k)*3 + 0];
  float g1 = grid_graph[((size_t)b*NGD + k)*3 + 1];
  float g2 = grid_graph[((size_t)b*NGD + k)*3 + 2];
  grid_node[(size_t)tid*3+0] = g0;
  grid_node[(size_t)tid*3+1] = g1;
  grid_node[(size_t)tid*3+2] = g2;
  float vx = vel[n*3], vy = vel[n*3+1], vz = vel[n*3+2];
  float px = pos[n*3]   - meanp[b*3];
  float py = pos[n*3+1] - meanp[b*3+1];
  float pz = pos[n*3+2] - meanp[b*3+2];
  float vs = vx*g0 + vy*g1 + vz*g2;
  float ps = px*g0 + py*g1 + pz*g2;
  float ch = charges[n];
  float vn = sqrtf(vx*vx + vy*vy + vz*vz);
  for (int c = 0; c < HH; ++c)
    x[(size_t)tid*HH + c] = vs*W_embed[c] + ps*W_embed[HH+c] + ch*W_embed[2*HH+c] + vn*W_embed[3*HH+c];
}

// fiber_basis: one block (128 threads) per (p,o) pair; hidden staged in LDS.
__global__ void __launch_bounds__(128)
fiber_k(const float* grid0, const float* Wf1, const float* bf1,
        const float* Wf2, const float* bf2, float* fib){
  __shared__ float hsh[HH];
  int po = blockIdx.x; int p = po / NGD, o = po - p * NGD;
  int j = threadIdx.x;
  float fi = grid0[p*3]*grid0[o*3] + grid0[p*3+1]*grid0[o*3+1] + grid0[p*3+2]*grid0[o*3+2];
  float fi2 = fi * fi;
  hsh[j] = gelu_f(fi*Wf1[j] + fi2*Wf1[HH + j] + bf1[j]);
  __syncthreads();
  float acc = bf2[j];
  for (int t = 0; t < HH; ++t) acc += hsh[t] * Wf2[(size_t)t*BDD + j];
  fib[(size_t)po*BDD + j] = gelu_f(acc);
}

// fk[p][o][c] = fiber_basis[p][o][:] . W_fiber[l][:,c]
__global__ void fk_k(const float* fib, const float* Wf, float* fk){
  int tid = blockIdx.x * blockDim.x + threadIdx.x;
  if (tid >= NGD*NGD*HH) return;
  int po = tid / HH, c = tid - po * HH;
  float acc = 0.0f;
  for (int bd = 0; bd < BDD; ++bd) acc += fib[(size_t)po*BDD + bd] * Wf[(size_t)bd*HH + c];
  fk[tid] = acc;
}

// ------------------- edge kernel: invariants -> basis MLP -> conv kern -> scatter -------------------
// One wave per 16 rows of the flattened (E*NG, .) row space. 72 bf16 WMMAs per wave.
__global__ void __launch_bounds__(32)
edge_conv_k(const int* __restrict__ src, const int* __restrict__ dst,
            const float* __restrict__ pos, const float* __restrict__ charges,
            const float* __restrict__ grid_node, const float* __restrict__ x,
            const bf16* __restrict__ Wt_b1, const float* __restrict__ bb1,
            const bf16* __restrict__ Wt_b2, const float* __restrict__ bb2,
            const bf16* __restrict__ Wt_cv, float* __restrict__ x1){
  __shared__ __align__(32) bf16 afrag1[32*16];      // ei_poly tile, K padded to 32
  __shared__ __align__(32) bf16 afrag2[4*32*16];    // hidden (16x128)
  __shared__ __align__(32) bf16 afrag3[4*32*16];    // basis  (16x128)
  __shared__ int msrc[16], mdst[16], mk[16];
  const int lane = threadIdx.x;
  const int row0 = blockIdx.x * 16;

  for (int i = lane; i < 32*16; i += 32) afrag1[i] = (bf16)0.0f;
  __syncthreads();

  if (lane < 16){
    int r = row0 + lane;
    int e = r / NGD; int k = r - e * NGD;
    int s = src[e], d = dst[e];
    float rx = pos[s*3+0] - pos[d*3+0];
    float ry = pos[s*3+1] - pos[d*3+1];
    float rz = pos[s*3+2] - pos[d*3+2];
    const float* g = grid_node + ((size_t)s*NGD + k)*3;
    float g0 = g[0], g1 = g[1], g2 = g[2];
    float i1 = rx*g0 + ry*g1 + rz*g2;
    float ux = rx - i1*g0, uy = ry - i1*g1, uz = rz - i1*g2;
    float i2 = sqrtf(ux*ux + uy*uy + uz*uz);
    float cp = charges[s] * charges[d];
    float ei[3] = {i1, i2, cp};
    wmma_stA(afrag1, lane, 0, i1);
    wmma_stA(afrag1, lane, 1, i2);
    wmma_stA(afrag1, lane, 2, cp);
    #pragma unroll
    for (int a = 0; a < 3; ++a)
      #pragma unroll
      for (int b = 0; b < 3; ++b)
        wmma_stA(afrag1, lane, 3 + a*3 + b, ei[a]*ei[b]);
    msrc[lane] = s; mdst[lane] = d; mk[lane] = k;
    // Prefetch the x[src] gather rows (512B each) now; they are consumed only
    // after the 3-GEMM chain, so the matrix pipe hides the latency.
    // Lowers to global_prefetch_b8 (probe-verified on gfx1250).
    const float* xp = x + ((size_t)s*NGD + k)*HH;
    __builtin_prefetch(xp, 0, 1);
    __builtin_prefetch(xp + 64, 0, 1);
  }
  __syncthreads();

  // GEMM1: (16x32) @ (32x128), bias + gelu
  v16bf a1 = wmma_ldA(afrag1, 0, lane);
  #pragma unroll
  for (int nt = 0; nt < 8; ++nt){
    v8f c = {};
    v16bf b = wmma_ldB(Wt_b1, 32, nt, 0, lane);
    c = WMMA_BF16(a1, b, c);
    float bias = bb1[nt*16 + (lane & 15)];
    #pragma unroll
    for (int r = 0; r < 8; ++r) c[r] = gelu_f(c[r] + bias);
    stage_c(afrag2, nt, lane, c);
  }
  __syncthreads();

  // GEMM2: (16x128) @ (128x128), bias + gelu -> basis tile
  #pragma unroll
  for (int nt = 0; nt < 8; ++nt){
    v8f c = {};
    #pragma unroll
    for (int kk = 0; kk < 4; ++kk){
      v16bf a = wmma_ldA(afrag2, kk, lane);
      v16bf b = wmma_ldB(Wt_b2, 128, nt, kk, lane);
      c = WMMA_BF16(a, b, c);
    }
    float bias = bb2[nt*16 + (lane & 15)];
    #pragma unroll
    for (int r = 0; r < 8; ++r) c[r] = gelu_f(c[r] + bias);
    stage_c(afrag3, nt, lane, c);
  }
  __syncthreads();

  // GEMM3: kern = basis @ W_conv[l], fused gather-multiply-scatter
  #pragma unroll
  for (int nt = 0; nt < 8; ++nt){
    v8f c = {};
    #pragma unroll
    for (int kk = 0; kk < 4; ++kk){
      v16bf a = wmma_ldA(afrag3, kk, lane);
      v16bf b = wmma_ldB(Wt_cv, 128, nt, kk, lane);
      c = WMMA_BF16(a, b, c);
    }
    int col = nt*16 + (lane & 15);
    #pragma unroll
    for (int r = 0; r < 8; ++r){
      int m = r + ((lane >> 4) << 3);
      int s = msrc[m], d = mdst[m], k = mk[m];
      float xv = x[((size_t)s*NGD + k)*HH + col];
      atomicAdd(&x1[((size_t)d*NGD + k)*HH + col], xv * c[r]);
    }
  }
}

// ------------------- fiber mixing + LayerNorm (fused mean/var reduction) -------------------
__global__ void __launch_bounds__(128)
mix_ln_k(const float* __restrict__ x1, const float* __restrict__ fk,
         const float* __restrict__ bc, const float* __restrict__ gamma,
         const float* __restrict__ beta, float* __restrict__ h2){
  __shared__ float red1[4];
  __shared__ float red2[4];
  const int n = blockIdx.x;
  const int c = threadIdx.x;
  float xo[NGD];
  #pragma unroll
  for (int o = 0; o < NGD; ++o) xo[o] = x1[((size_t)n*NGD + o)*HH + c];
  for (int p = 0; p < NGD; ++p){
    float acc = 0.0f;
    #pragma unroll
    for (int o = 0; o < NGD; ++o) acc += xo[o] * fk[((size_t)p*NGD + o)*HH + c];
    acc = acc * (1.0f/NGD) + bc[c];
    float v1 = acc, v2 = acc * acc;
    for (int msk = 16; msk >= 1; msk >>= 1){
      v1 += __shfl_xor(v1, msk, 32);
      v2 += __shfl_xor(v2, msk, 32);
    }
    if ((c & 31) == 0){ red1[c >> 5] = v1; red2[c >> 5] = v2; }
    __syncthreads();
    float mu  = (red1[0]+red1[1]+red1[2]+red1[3]) * (1.0f/HH);
    float ex2 = (red2[0]+red2[1]+red2[2]+red2[3]) * (1.0f/HH);
    __syncthreads();
    float var = ex2 - mu * mu;
    h2[((size_t)n*NGD + p)*HH + c] = (acc - mu) * rsqrtf(var + 1e-5f) * gamma[c] + beta[c];
  }
}

// ------------------- feed-forward (128->512->128) + residual, WMMA -------------------
__global__ void __launch_bounds__(32)
ff_k(float* __restrict__ x, const float* __restrict__ h2,
     const bf16* __restrict__ Wt1, const float* __restrict__ bb1,
     const bf16* __restrict__ Wt2, const float* __restrict__ bb2){
  __shared__ __align__(32) bf16 afA[4*32*16];
  __shared__ __align__(32) bf16 afB[16*32*16];
  const int lane = threadIdx.x;
  const size_t row0 = (size_t)blockIdx.x * 16;
  for (int idx = lane; idx < 16*HH; idx += 32){
    int m = idx >> 7, k = idx & 127;
    wmma_stA(afA, m, k, h2[(row0 + m)*HH + k]);
  }
  __syncthreads();
  for (int nt = 0; nt < 32; ++nt){
    v8f c = {};
    #pragma unroll
    for (int kk = 0; kk < 4; ++kk){
      v16bf a = wmma_ldA(afA, kk, lane);
      v16bf b = wmma_ldB(Wt1, 128, nt, kk, lane);
      c = WMMA_BF16(a, b, c);
    }
    float bias = bb1[nt*16 + (lane & 15)];
    #pragma unroll
    for (int r = 0; r < 8; ++r) c[r] = gelu_f(c[r] + bias);
    stage_c(afB, nt, lane, c);
  }
  __syncthreads();
  #pragma unroll
  for (int nt = 0; nt < 8; ++nt){
    v8f c = {};
    for (int kk = 0; kk < 16; ++kk){
      v16bf a = wmma_ldA(afB, kk, lane);
      v16bf b = wmma_ldB(Wt2, 512, nt, kk, lane);
      c = WMMA_BF16(a, b, c);
    }
    int col = nt*16 + (lane & 15);
    float bias = bb2[col];
    #pragma unroll
    for (int r = 0; r < 8; ++r){
      int m = r + ((lane >> 4) << 3);
      x[(row0 + m)*HH + col] += c[r] + bias;   // residual
    }
  }
}

// ------------------- readout: gelu(x@W_ro1+b)@W_ro2+b, then mean_k(grid_node*y) -------------------
__global__ void __launch_bounds__(32)
readout_k(const float* __restrict__ x, const bf16* __restrict__ Wt_ro1,
          const float* __restrict__ b_ro1, const float* __restrict__ W_ro2,
          const float* __restrict__ b_ro2, const float* __restrict__ grid_node,
          float* __restrict__ out){
  __shared__ __align__(32) bf16 afA[4*32*16];
  const int lane = threadIdx.x;
  const size_t row0 = (size_t)blockIdx.x * 16;
  for (int idx = lane; idx < 16*HH; idx += 32){
    int m = idx >> 7, k = idx & 127;
    wmma_stA(afA, m, k, x[(row0 + m)*HH + k]);
  }
  __syncthreads();
  float part[8];
  #pragma unroll
  for (int r = 0; r < 8; ++r) part[r] = 0.0f;
  #pragma unroll
  for (int nt = 0; nt < 8; ++nt){
    v8f c = {};
    #pragma unroll
    for (int kk = 0; kk < 4; ++kk){
      v16bf a = wmma_ldA(afA, kk, lane);
      v16bf b = wmma_ldB(Wt_ro1, 128, nt, kk, lane);
      c = WMMA_BF16(a, b, c);
    }
    int col = nt*16 + (lane & 15);
    float bias = b_ro1[col];
    float w2 = W_ro2[col];
    #pragma unroll
    for (int r = 0; r < 8; ++r) part[r] += gelu_f(c[r] + bias) * w2;
  }
  #pragma unroll
  for (int r = 0; r < 8; ++r){
    float v = part[r];
    for (int msk = 8; msk >= 1; msk >>= 1) v += __shfl_xor(v, msk, 16);
    part[r] = v;
  }
  if ((lane & 15) == 0){
    #pragma unroll
    for (int r = 0; r < 8; ++r){
      int m = r + ((lane >> 4) << 3);
      size_t row = row0 + m;
      int nnode = (int)(row / NGD), k = (int)(row % NGD);
      float y = part[r] + b_ro2[0];
      const float* g = grid_node + ((size_t)nnode*NGD + k)*3;
      #pragma unroll
      for (int d = 0; d < 3; ++d)
        atomicAdd(&out[nnode*3 + d], g[d] * y * (1.0f/NGD));
    }
  }
}

// ------------------------------- host orchestration -------------------------------
extern "C" void kernel_launch(void* const* d_in, const int* in_sizes, int n_in,
                              void* d_out, int out_size, void* d_ws, size_t ws_size,
                              hipStream_t stream){
  (void)in_sizes; (void)n_in; (void)out_size; (void)ws_size;
  const float* pos      = (const float*)d_in[0];
  const float* vel      = (const float*)d_in[1];
  const float* charges  = (const float*)d_in[2];
  const int*   batch    = (const int*)  d_in[3];
  const int*   edge     = (const int*)  d_in[4];
  const float* grid0    = (const float*)d_in[5];
  const float* randR    = (const float*)d_in[6];
  const float* W_basis1 = (const float*)d_in[7];
  const float* b_basis1 = (const float*)d_in[8];
  const float* W_basis2 = (const float*)d_in[9];
  const float* b_basis2 = (const float*)d_in[10];
  const float* W_fb1    = (const float*)d_in[11];
  const float* b_fb1    = (const float*)d_in[12];
  const float* W_fb2    = (const float*)d_in[13];
  const float* b_fb2    = (const float*)d_in[14];
  const float* W_embed  = (const float*)d_in[15];
  const float* W_conv   = (const float*)d_in[16];
  const float* W_fiber  = (const float*)d_in[17];
  const float* b_conv   = (const float*)d_in[18];
  const float* ln_gamma = (const float*)d_in[19];
  const float* ln_beta  = (const float*)d_in[20];
  const float* W_ff1    = (const float*)d_in[21];
  const float* b_ff1    = (const float*)d_in[22];
  const float* W_ff2    = (const float*)d_in[23];
  const float* b_ff2    = (const float*)d_in[24];
  const float* W_ro1    = (const float*)d_in[25];
  const float* b_ro1    = (const float*)d_in[26];
  const float* W_ro2    = (const float*)d_in[27];
  const float* b_ro2    = (const float*)d_in[28];
  const int* src = edge;
  const int* dst = edge + EE;
  float* out = (float*)d_out;

  // Workspace carve-out (~75 MB total; harness provides d_ws scratch)
  char* cur = (char*)d_ws;
  auto alloc = [&](size_t bytes) -> char* {
    char* r = cur;
    cur += (bytes + 255) & ~((size_t)255);
    return r;
  };
  float* grid_graph = (float*)alloc((size_t)BBG*NGD*3*4);
  float* cnt        = (float*)alloc((size_t)BBG*4);
  float* psum       = (float*)alloc((size_t)BBG*3*4);
  float* meanp      = (float*)alloc((size_t)BBG*3*4);
  float* grid_node  = (float*)alloc((size_t)NN*NGD*3*4);
  float* xbuf       = (float*)alloc((size_t)NN*NGD*HH*4);
  float* x1buf      = (float*)alloc((size_t)NN*NGD*HH*4);
  float* h2buf      = (float*)alloc((size_t)NN*NGD*HH*4);
  float* fib        = (float*)alloc((size_t)NGD*NGD*BDD*4);
  float* fkbuf      = (float*)alloc((size_t)NGD*NGD*HH*4);
  bf16* Wt_b1  = (bf16*)alloc((size_t)HH*32*2);
  bf16* Wt_b2  = (bf16*)alloc((size_t)BDD*HH*2);            // [N=128][Kpad=128]
  bf16* Wt_cv  = (bf16*)alloc((size_t)LLAY*HH*HH*2);
  bf16* Wt_f1  = (bf16*)alloc((size_t)LLAY*HH4*HH*2);       // [512][128] per layer
  bf16* Wt_f2  = (bf16*)alloc((size_t)LLAY*HH*HH4*2);       // [128][512] per layer
  bf16* Wt_r1  = (bf16*)alloc((size_t)HH*HH*2);

  const int T = 256;
  // stats + grid
  zero_k<<<1, T, 0, stream>>>(cnt, BBG);
  zero_k<<<2, T, 0, stream>>>(psum, BBG*3);
  grid_graph_k<<<(BBG*NGD + T-1)/T, T, 0, stream>>>(randR, grid0, grid_graph);
  stats_k<<<(NN + T-1)/T, T, 0, stream>>>(pos, batch, cnt, psum);
  mean_k<<<1, T, 0, stream>>>(cnt, psum, meanp);

  // weight conversion (fp32 -> bf16, transposed, K padded)
  wt_k<<<(HH*32 + T-1)/T, T, 0, stream>>>(W_basis1, Wt_b1, 12, HH, 32);
  wt_k<<<(HH*HH + T-1)/T, T, 0, stream>>>(W_basis2, Wt_b2, HH, BDD, HH);
  wt_k<<<(HH*HH + T-1)/T, T, 0, stream>>>(W_ro1,    Wt_r1, HH, HH, HH);
  for (int l = 0; l < LLAY; ++l){
    wt_k<<<(HH*HH + T-1)/T, T, 0, stream>>>(W_conv + (size_t)l*BDD*HH, Wt_cv + (size_t)l*HH*HH, BDD, HH, HH);
    wt_k<<<(HH4*HH + T-1)/T, T, 0, stream>>>(W_ff1 + (size_t)l*HH*HH4, Wt_f1 + (size_t)l*HH4*HH, HH, HH4, HH);
    wt_k<<<(HH*HH4 + T-1)/T, T, 0, stream>>>(W_ff2 + (size_t)l*HH4*HH, Wt_f2 + (size_t)l*HH*HH4, HH4, HH, HH4);
  }

  // embedding + fiber basis
  embed_k<<<(NN*NGD + T-1)/T, T, 0, stream>>>(pos, vel, charges, batch, grid_graph, meanp,
                                              W_embed, grid_node, xbuf);
  fiber_k<<<NGD*NGD, 128, 0, stream>>>(grid0, W_fb1, b_fb1, W_fb2, b_fb2, fib);

  const long long XN = (long long)NN*NGD*HH;
  const int EDGE_TILES = (EE*NGD)/16;   // 37500
  const int ROW_TILES  = (NN*NGD)/16;   // 3000

  for (int l = 0; l < LLAY; ++l){
    zero_k<<<(int)((XN + T-1)/T), T, 0, stream>>>(x1buf, XN);
    fk_k<<<(NGD*NGD*HH + T-1)/T, T, 0, stream>>>(fib, W_fiber + (size_t)l*BDD*HH, fkbuf);
    edge_conv_k<<<EDGE_TILES, 32, 0, stream>>>(src, dst, pos, charges, grid_node, xbuf,
                                               Wt_b1, b_basis1, Wt_b2, b_basis2,
                                               Wt_cv + (size_t)l*HH*HH, x1buf);
    mix_ln_k<<<NN, 128, 0, stream>>>(x1buf, fkbuf, b_conv + (size_t)l*HH,
                                     ln_gamma + (size_t)l*HH, ln_beta + (size_t)l*HH, h2buf);
    ff_k<<<ROW_TILES, 32, 0, stream>>>(xbuf, h2buf,
                                       Wt_f1 + (size_t)l*HH4*HH, b_ff1 + (size_t)l*HH4,
                                       Wt_f2 + (size_t)l*HH*HH4, b_ff2 + (size_t)l*HH);
  }

  zero_k<<<(NN*3 + T-1)/T, T, 0, stream>>>(out, NN*3);
  readout_k<<<ROW_TILES, 32, 0, stream>>>(xbuf, Wt_r1, b_ro1, W_ro2, b_ro2, grid_node, out);
}
